// DPPNMT_66202625900878
// MI455X (gfx1250) — matrix operand order repaired and checked by reading, not compiled, not run on hardware
//
#include <hip/hip_runtime.h>
#include <math.h>

// ---------------------------------------------------------------------------
// NMT forward (bi-LSTM encoder + attention decoder + vocab log-softmax gather)
// for MI455X / gfx1250 (wave32, WMMA f32<-f16 16x16x32).
//
// S=T=64, B=64, E=H=256, 4H=1024, V=32000.
// All GEMMs use v_wmma_f32_16x16x32_f16 (f16 inputs, f32 accumulation).
// Input-side LSTM GEMMs are hoisted out of the recurrences; the vocab
// projection streams a per-row logsumexp instead of materializing logits.
// ---------------------------------------------------------------------------

typedef __attribute__((ext_vector_type(16))) _Float16 v16h;
typedef __attribute__((ext_vector_type(8)))  _Float16 v8h;
typedef __attribute__((ext_vector_type(8)))  float    v8f;

#define NS   64      // source length
#define NT   64      // target length
#define NB   64      // batch
#define NE   256     // embed
#define NH   256     // hidden
#define NH4  1024    // 4*H
#define NV   32000   // vocab
#define NT1  63      // T-1
#define MVOC (NT1*NB)

__device__ __forceinline__ float sigf(float x) { return 1.0f / (1.0f + expf(-x)); }

// ---------------------------------------------------------------------------
// Generic WMMA GEMM:  C[M,N] = A[M,K](f16) @ W[N,K]^T(f16) (+Cin) (+bias)
// Block = 128 threads = 4 waves; wave w computes rows [by*64+w*16, +16),
// cols [bx*64, +64) as 4 accumulators of 16x16. M,N multiples of 64; K of 32.
// ---------------------------------------------------------------------------
__global__ __launch_bounds__(128)
void nmt_gemm(const _Float16* __restrict__ A, int lda,
              const _Float16* __restrict__ W, int ldw,
              const float* __restrict__ Cin,
              const float* __restrict__ bias,
              float* __restrict__ C,
              int N, int K) {
  const int wave  = threadIdx.x >> 5;
  const int lane  = threadIdx.x & 31;
  const int lhalf = (lane >= 16) ? 1 : 0;
  const int l16   = lane & 15;
  const int row0  = blockIdx.y * 64 + wave * 16;
  const int col0  = blockIdx.x * 64;

  v8f acc[4] = {};
  const _Float16* arow = A + (size_t)(row0 + l16) * lda;

  for (int k0 = 0; k0 < K; k0 += 32) {
    // A fragment (16x32 f16): lane<16 holds K {k0..k0+7, k0+16..k0+23},
    // lane>=16 holds K {k0+8..k0+15, k0+24..k0+31}; M = lane%16.
    const v8h alo = *(const v8h*)(arow + k0 + (lhalf ? 8 : 0));
    const v8h ahi = *(const v8h*)(arow + k0 + (lhalf ? 8 : 0) + 16);
    v16h a;
#pragma unroll
    for (int i = 0; i < 8; ++i) { a[i] = alo[i]; a[i + 8] = ahi[i]; }
#pragma unroll
    for (int j = 0; j < 4; ++j) {
      // B fragment (32x16): N = col + lane%16; same K split as A:
      // lane<16 K {k0..k0+7, k0+16..k0+23}, lane>=16 K {k0+8..k0+15, k0+24..k0+31}.
      const _Float16* wp = W + (size_t)(col0 + j * 16 + l16) * ldw + k0 + (lhalf ? 8 : 0);
      const v8h blo = *(const v8h*)(wp);
      const v8h bhi = *(const v8h*)(wp + 16);
      v16h b;
#pragma unroll
      for (int i = 0; i < 8; ++i) { b[i] = blo[i]; b[i + 8] = bhi[i]; }
      acc[j] = __builtin_amdgcn_wmma_f32_16x16x32_f16(
          false, a, false, b, (short)0, acc[j], false, false);
    }
  }

#pragma unroll
  for (int j = 0; j < 4; ++j) {
    const int col = col0 + j * 16 + l16;
    const float bv = bias ? bias[col] : 0.0f;
#pragma unroll
    for (int r = 0; r < 8; ++r) {
      const int row = row0 + r + (lhalf ? 8 : 0);
      const size_t idx = (size_t)row * N + col;
      float v = acc[j][r] + bv;
      if (Cin) v += Cin[idx];
      C[idx] = v;
    }
  }
}

// ---------------------------------------------------------------------------
// Vocab GEMM with streaming logsumexp epilogue: never materializes logits.
// Per block tile (64 rows x 64 cols) each wave reduces its 16 rows over the
// 64 columns to (max, sumexp) and writes one partial per (row, col-chunk).
// ---------------------------------------------------------------------------
__global__ __launch_bounds__(128)
void nmt_gemm_lse(const _Float16* __restrict__ A, int lda,
                  const _Float16* __restrict__ W, int ldw,
                  float2* __restrict__ partials, int K) {
  const int wave  = threadIdx.x >> 5;
  const int lane  = threadIdx.x & 31;
  const int lhalf = (lane >= 16) ? 1 : 0;
  const int l16   = lane & 15;
  const int row0  = blockIdx.y * 64 + wave * 16;
  const int col0  = blockIdx.x * 64;
  const int nch   = gridDim.x;

  v8f acc[4] = {};
  const _Float16* arow = A + (size_t)(row0 + l16) * lda;

  for (int k0 = 0; k0 < K; k0 += 32) {
    const v8h alo = *(const v8h*)(arow + k0 + (lhalf ? 8 : 0));
    const v8h ahi = *(const v8h*)(arow + k0 + (lhalf ? 8 : 0) + 16);
    v16h a;
#pragma unroll
    for (int i = 0; i < 8; ++i) { a[i] = alo[i]; a[i + 8] = ahi[i]; }
#pragma unroll
    for (int j = 0; j < 4; ++j) {
      const _Float16* wp = W + (size_t)(col0 + j * 16 + l16) * ldw + k0 + (lhalf ? 8 : 0);
      const v8h blo = *(const v8h*)(wp);
      const v8h bhi = *(const v8h*)(wp + 16);
      v16h b;
#pragma unroll
      for (int i = 0; i < 8; ++i) { b[i] = blo[i]; b[i + 8] = bhi[i]; }
      acc[j] = __builtin_amdgcn_wmma_f32_16x16x32_f16(
          false, a, false, b, (short)0, acc[j], false, false);
    }
  }

  // Per-row (max, sum-exp) across this block's 64 columns.
  // Row (row0 + r + 8*lhalf) lives in VGPR r of the 16 lanes of one half;
  // xor-shuffles with masks 8..1 stay within the 16-lane half.
#pragma unroll
  for (int r = 0; r < 8; ++r) {
    float m = fmaxf(fmaxf(acc[0][r], acc[1][r]), fmaxf(acc[2][r], acc[3][r]));
    for (int d = 8; d >= 1; d >>= 1) m = fmaxf(m, __shfl_xor(m, d));
    float s = 0.0f;
#pragma unroll
    for (int j = 0; j < 4; ++j) s += expf(acc[j][r] - m);
    for (int d = 8; d >= 1; d >>= 1) s += __shfl_xor(s, d);
    if (l16 == 0) {
      const int row = row0 + r + (lhalf ? 8 : 0);
      partials[(size_t)row * nch + blockIdx.x] = make_float2(m, s);
    }
  }
}

__device__ __forceinline__ void lse_merge(float& m, float& s, float m2, float s2) {
  const float mn = fmaxf(m, m2);
  s = s * expf(m - mn) + s2 * expf(m2 - mn);
  m = mn;
}

__global__ void nmt_lse_reduce(const float2* __restrict__ partials, int nch,
                               float* __restrict__ lse) {
  const int row  = blockIdx.x;
  const int lane = threadIdx.x;
  float m = -3.0e38f, s = 0.0f;
  for (int c = lane; c < nch; c += 32) {
    const float2 p = partials[(size_t)row * nch + c];
    lse_merge(m, s, p.x, p.y);
  }
  for (int d = 16; d >= 1; d >>= 1) {
    const float m2 = __shfl_xor(m, d);
    const float s2 = __shfl_xor(s, d);
    lse_merge(m, s, m2, s2);
  }
  if (lane == 0) lse[row] = m + logf(s);
}

// Gold logit: dot(outs_h[row,:], Wvocab_h[token,:]) over H, one wave per row.
__global__ void nmt_gold(const _Float16* __restrict__ outs_h,
                         const _Float16* __restrict__ wvoc,
                         const int* __restrict__ tgt,
                         float* __restrict__ gold) {
  const int row  = blockIdx.x;
  const int lane = threadIdx.x;
  const int t = row / NB, b = row - t * NB;
  const int tok = tgt[(t + 1) * NB + b];
  const _Float16* x = outs_h + (size_t)row * NH;
  const _Float16* w = wvoc + (size_t)tok * NH;
  float s = 0.0f;
  for (int k = lane; k < NH; k += 32) s += (float)x[k] * (float)w[k];
  for (int d = 16; d >= 1; d >>= 1) s += __shfl_xor(s, d);
  if (lane == 0) gold[row] = s;
}

__global__ void nmt_final(const float* __restrict__ gold,
                          const float* __restrict__ lse,
                          const int* __restrict__ tgt,
                          float* __restrict__ out) {
  const int b = threadIdx.x;
  float acc = 0.0f;
  for (int t = 0; t < NT1; ++t) {
    const int tok = tgt[(t + 1) * NB + b];
    if (tok != 0) acc += gold[t * NB + b] - lse[t * NB + b];  // PAD == 0
  }
  out[b] = acc;
}

// ------------------------- small helper kernels ----------------------------
__global__ void nmt_f2h(const float* __restrict__ in, _Float16* __restrict__ out, int n) {
  const int i = blockIdx.x * blockDim.x + threadIdx.x;
  if (i < n) out[i] = (_Float16)in[i];
}
__global__ void nmt_zero_h(_Float16* __restrict__ p, int n) {
  const int i = blockIdx.x * blockDim.x + threadIdx.x;
  if (i < n) p[i] = (_Float16)0.0f;
}
__global__ void nmt_zero_f(float* __restrict__ p, int n) {
  const int i = blockIdx.x * blockDim.x + threadIdx.x;
  if (i < n) p[i] = 0.0f;
}
__global__ void nmt_gather(const int* __restrict__ idx, const float* __restrict__ emb,
                           _Float16* __restrict__ out, int n) {  // n = rows*NE
  const int i = blockIdx.x * blockDim.x + threadIdx.x;
  if (i >= n) return;
  const int r = i / NE, e = i - r * NE;
  out[i] = (_Float16)emb[(size_t)idx[r] * NE + e];
}
__global__ void nmt_concat_hh(const _Float16* __restrict__ x, const _Float16* __restrict__ y,
                              _Float16* __restrict__ out) {  // (B,H)+(B,H)->(B,2H)
  const int i = blockIdx.x * blockDim.x + threadIdx.x;      // B*2H
  const int b = i / (2 * NH), j = i - b * 2 * NH;
  out[i] = (j < NH) ? x[b * NH + j] : y[b * NH + (j - NH)];
}
__global__ void nmt_concat_ff2h(const float* __restrict__ x, const float* __restrict__ y,
                                _Float16* __restrict__ out) {
  const int i = blockIdx.x * blockDim.x + threadIdx.x;
  const int b = i / (2 * NH), j = i - b * 2 * NH;
  out[i] = (_Float16)((j < NH) ? x[b * NH + j] : y[b * NH + (j - NH)]);
}

// LSTM gate update: z(B,4H) [i|f|g|o], c in/out (B,H), writes h (f32+f16),
// and optionally a strided extra f16 copy (used to build enc_hiddens).
__global__ void nmt_lstm_gate(const float* __restrict__ z,
                              float* __restrict__ c,
                              float* __restrict__ hF,
                              _Float16* __restrict__ hH,
                              _Float16* __restrict__ extra, int extraRowStride) {
  const int i = blockIdx.x * blockDim.x + threadIdx.x;  // B*H
  const int b = i / NH, j = i - b * NH;
  const float* zb = z + (size_t)b * NH4;
  const float ig = sigf(zb[j]);
  const float fg = sigf(zb[NH + j]);
  const float gg = tanhf(zb[2 * NH + j]);
  const float og = sigf(zb[3 * NH + j]);
  const float cn = fg * c[i] + ig * gg;
  c[i] = cn;
  const float hn = og * tanhf(cn);
  hF[i] = hn;
  hH[i] = (_Float16)hn;
  if (extra) extra[(size_t)b * extraRowStride + j] = (_Float16)hn;
}

// e[b,s] = dot(enc_proj[b,s,:], h[b,:]) over H.
__global__ void nmt_et(const float* __restrict__ enc_proj, const float* __restrict__ hF,
                       float* __restrict__ e) {
  const int i = blockIdx.x * blockDim.x + threadIdx.x;  // B*S
  if (i >= NB * NS) return;
  const int b = i / NS;
  const float* p = enc_proj + (size_t)i * NH;
  const float* h = hF + (size_t)b * NH;
  float s = 0.0f;
  for (int k = 0; k < NH; ++k) s += p[k] * h[k];
  e[i] = s;
}

// Row softmax over S=64: one wave per batch row, 2 elements per lane.
__global__ void nmt_softmax(const float* __restrict__ e, float* __restrict__ alpha) {
  const int b = blockIdx.x, lane = threadIdx.x;
  const float v0 = e[b * NS + lane];
  const float v1 = e[b * NS + lane + 32];
  float mx = fmaxf(v0, v1);
  for (int d = 16; d >= 1; d >>= 1) mx = fmaxf(mx, __shfl_xor(mx, d));
  const float e0 = expf(v0 - mx), e1 = expf(v1 - mx);
  float s = e0 + e1;
  for (int d = 16; d >= 1; d >>= 1) s += __shfl_xor(s, d);
  alpha[b * NS + lane]      = e0 / s;
  alpha[b * NS + lane + 32] = e1 / s;
}

// a_t = alpha @ enc_hiddens, packed with h into u = [a_t(2H) | h(H)] (f16).
__global__ void nmt_att(const float* __restrict__ alpha, const _Float16* __restrict__ enc_h,
                        const _Float16* __restrict__ hH, _Float16* __restrict__ u) {
  const int i = blockIdx.x * blockDim.x + threadIdx.x;  // B*3H
  const int b = i / (3 * NH), d = i - b * 3 * NH;
  if (d < 2 * NH) {
    float s = 0.0f;
    const float* al = alpha + b * NS;
    const _Float16* eh = enc_h + (size_t)b * NS * 2 * NH + d;
    for (int t = 0; t < NS; ++t) s += al[t] * (float)eh[(size_t)t * 2 * NH];
    u[i] = (_Float16)s;
  } else {
    u[i] = hH[b * NH + (d - 2 * NH)];
  }
}

__global__ void nmt_tanh_h(const float* __restrict__ in, _Float16* __restrict__ out, int n) {
  const int i = blockIdx.x * blockDim.x + threadIdx.x;
  if (i < n) out[i] = (_Float16)tanhf(in[i]);
}

// ---------------------------------------------------------------------------
extern "C" void kernel_launch(void* const* d_in, const int* in_sizes, int n_in,
                              void* d_out, int out_size, void* d_ws, size_t ws_size,
                              hipStream_t stream) {
  (void)in_sizes; (void)n_in; (void)out_size; (void)ws_size;

  const int*   src_pad  = (const int*)d_in[0];
  const int*   tgt_pad  = (const int*)d_in[1];
  const float* src_emb  = (const float*)d_in[2];
  const float* tgt_emb  = (const float*)d_in[3];
  const float* eWih_f   = (const float*)d_in[4];
  const float* eWhh_f   = (const float*)d_in[5];
  const float* eb_f     = (const float*)d_in[6];
  const float* eWih_b   = (const float*)d_in[7];
  const float* eWhh_b   = (const float*)d_in[8];
  const float* eb_b     = (const float*)d_in[9];
  const float* dWih     = (const float*)d_in[10];
  const float* dWhh     = (const float*)d_in[11];
  const float* db       = (const float*)d_in[12];
  const float* Wh       = (const float*)d_in[13];
  const float* Wc       = (const float*)d_in[14];
  const float* Watt     = (const float*)d_in[15];
  const float* Wcomb    = (const float*)d_in[16];
  const float* Wvocab   = (const float*)d_in[17];
  float* out = (float*)d_out;

  // ---- workspace bump allocator (aligned 256B) ----
  char* base = (char*)d_ws;
  size_t off = 0;
  auto alloc = [&](size_t bytes) -> void* {
    void* p = base + off;
    off = (off + bytes + 255) & ~(size_t)255;
    return p;
  };
  auto allocH = [&](size_t n) -> _Float16* { return (_Float16*)alloc(n * sizeof(_Float16)); };
  auto allocF = [&](size_t n) -> float*    { return (float*)alloc(n * sizeof(float)); };

  // f16 weights
  _Float16* hWih_f  = allocH(NH4 * NE);
  _Float16* hWhh_f  = allocH(NH4 * NH);
  _Float16* hWih_b  = allocH(NH4 * NE);
  _Float16* hWhh_b  = allocH(NH4 * NH);
  _Float16* hdWih   = allocH(NH4 * (NE + NH));
  _Float16* hdWhh   = allocH(NH4 * NH);
  _Float16* hWh     = allocH(NH * 2 * NH);
  _Float16* hWc     = allocH(NH * 2 * NH);
  _Float16* hWatt   = allocH(NH * 2 * NH);
  _Float16* hWcomb  = allocH(NH * 3 * NH);
  _Float16* hWvoc   = allocH((size_t)NV * NH);
  // f16 activations
  _Float16* Xsrc_h  = allocH((size_t)NS * NB * NE);
  _Float16* Y_h     = allocH((size_t)NT1 * NB * NE);
  _Float16* enc_h   = allocH((size_t)NB * NS * 2 * NH);
  _Float16* hA_h    = allocH(NB * NH);
  _Float16* hB_h    = allocH(NB * NH);
  _Float16* cat_h   = allocH(NB * 2 * NH);
  _Float16* catc_h  = allocH(NB * 2 * NH);
  _Float16* hdec_h  = allocH(NB * NH);
  _Float16* u_h     = allocH(NB * 3 * NH);
  _Float16* oprev0  = allocH(NB * NH);
  _Float16* outs_h  = allocH((size_t)NT1 * NB * NH);
  // f32 buffers
  float* Zx_f   = allocF((size_t)NS * NB * NH4);
  float* Zx_b   = allocF((size_t)NS * NB * NH4);
  float* Zy     = allocF((size_t)NT1 * NB * NH4);
  float* zbuf   = allocF(NB * NH4);
  float* cA     = allocF(NB * NH);
  float* cB     = allocF(NB * NH);
  float* c_dec  = allocF(NB * NH);
  float* hF     = allocF(NB * NH);
  float* e_proj = allocF((size_t)NB * NS * NH);
  float* e_buf  = allocF(NB * NS);
  float* alpha  = allocF(NB * NS);
  float* Opre   = allocF(NB * NH);
  float* lse    = allocF(MVOC);
  float* gold   = allocF(MVOC);
  const int NCH = NV / 64;  // 500 column chunks for vocab GEMM
  float2* parts = (float2*)alloc((size_t)MVOC * NCH * sizeof(float2));

  // ---- 1. convert weights to f16 ----
  struct Cvt { const float* s; _Float16* d; int n; };
  const Cvt cv[] = {
      {eWih_f, hWih_f, NH4 * NE}, {eWhh_f, hWhh_f, NH4 * NH},
      {eWih_b, hWih_b, NH4 * NE}, {eWhh_b, hWhh_b, NH4 * NH},
      {dWih, hdWih, NH4 * (NE + NH)}, {dWhh, hdWhh, NH4 * NH},
      {Wh, hWh, NH * 2 * NH}, {Wc, hWc, NH * 2 * NH},
      {Watt, hWatt, NH * 2 * NH}, {Wcomb, hWcomb, NH * 3 * NH},
      {Wvocab, hWvoc, NV * NH}};
  for (const auto& c : cv)
    nmt_f2h<<<(c.n + 255) / 256, 256, 0, stream>>>(c.s, c.d, c.n);

  // ---- 2. embedding gathers (f32 -> f16) ----
  nmt_gather<<<(NS * NB * NE) / 256, 256, 0, stream>>>(src_pad, src_emb, Xsrc_h, NS * NB * NE);
  nmt_gather<<<(NT1 * NB * NE) / 256, 256, 0, stream>>>(tgt_pad, tgt_emb, Y_h, NT1 * NB * NE);

  // ---- 3. hoisted input GEMMs: Zx = X@Wih^T + b (both dirs), Zy = Y@dWih[:, :E]^T + db
  nmt_gemm<<<dim3(NH4 / 64, (NS * NB) / 64), 128, 0, stream>>>(
      Xsrc_h, NE, hWih_f, NE, nullptr, eb_f, Zx_f, NH4, NE);
  nmt_gemm<<<dim3(NH4 / 64, (NS * NB) / 64), 128, 0, stream>>>(
      Xsrc_h, NE, hWih_b, NE, nullptr, eb_b, Zx_b, NH4, NE);
  nmt_gemm<<<dim3(NH4 / 64, (NT1 * NB) / 64), 128, 0, stream>>>(
      Y_h, NE, hdWih, NE + NH, nullptr, db, Zy, NH4, NE);

  // ---- 4. forward encoder LSTM ----
  nmt_zero_h<<<(NB * NH) / 256, 256, 0, stream>>>(hA_h, NB * NH);
  nmt_zero_f<<<(NB * NH) / 256, 256, 0, stream>>>(cA, NB * NH);
  for (int t = 0; t < NS; ++t) {
    nmt_gemm<<<dim3(NH4 / 64, 1), 128, 0, stream>>>(
        hA_h, NH, hWhh_f, NH, Zx_f + (size_t)t * NB * NH4, nullptr, zbuf, NH4, NH);
    nmt_lstm_gate<<<(NB * NH) / 256, 256, 0, stream>>>(
        zbuf, cA, hF, hA_h, enc_h + (size_t)t * 2 * NH, NS * 2 * NH);
  }
  // ---- 5. backward encoder LSTM (time reversed, written back in place) ----
  nmt_zero_h<<<(NB * NH) / 256, 256, 0, stream>>>(hB_h, NB * NH);
  nmt_zero_f<<<(NB * NH) / 256, 256, 0, stream>>>(cB, NB * NH);
  for (int s = 0; s < NS; ++s) {
    const int t = NS - 1 - s;
    nmt_gemm<<<dim3(NH4 / 64, 1), 128, 0, stream>>>(
        hB_h, NH, hWhh_b, NH, Zx_b + (size_t)t * NB * NH4, nullptr, zbuf, NH4, NH);
    nmt_lstm_gate<<<(NB * NH) / 256, 256, 0, stream>>>(
        zbuf, cB, hF, hB_h, enc_h + (size_t)t * 2 * NH + NH, NS * 2 * NH);
  }

  // ---- 6. decoder init: h0 = [hf|hb]@Wh^T, c0 = [cf|cb]@Wc^T ----
  nmt_concat_hh<<<(NB * 2 * NH) / 256, 256, 0, stream>>>(hA_h, hB_h, cat_h);
  nmt_concat_ff2h<<<(NB * 2 * NH) / 256, 256, 0, stream>>>(cA, cB, catc_h);
  nmt_gemm<<<dim3(NH / 64, 1), 128, 0, stream>>>(
      cat_h, 2 * NH, hWh, 2 * NH, nullptr, nullptr, hF, NH, 2 * NH);
  nmt_f2h<<<(NB * NH) / 256, 256, 0, stream>>>(hF, hdec_h, NB * NH);
  nmt_gemm<<<dim3(NH / 64, 1), 128, 0, stream>>>(
      catc_h, 2 * NH, hWc, 2 * NH, nullptr, nullptr, c_dec, NH, 2 * NH);

  // ---- 7. enc_proj = enc_hiddens @ Watt^T  (M = B*S) ----
  nmt_gemm<<<dim3(NH / 64, (NB * NS) / 64), 128, 0, stream>>>(
      enc_h, 2 * NH, hWatt, 2 * NH, nullptr, nullptr, e_proj, NH, 2 * NH);

  // ---- 8. decoder loop ----
  nmt_zero_h<<<(NB * NH) / 256, 256, 0, stream>>>(oprev0, NB * NH);
  for (int t = 0; t < NT1; ++t) {
    const _Float16* oprev = (t == 0) ? oprev0 : (outs_h + (size_t)(t - 1) * NB * NH);
    // z = Zy[t] + o_prev @ dWih[:, E:]^T
    nmt_gemm<<<dim3(NH4 / 64, 1), 128, 0, stream>>>(
        oprev, NH, hdWih + NE, NE + NH, Zy + (size_t)t * NB * NH4, nullptr, zbuf, NH4, NH);
    // z += h @ dWhh^T (in-place accumulate)
    nmt_gemm<<<dim3(NH4 / 64, 1), 128, 0, stream>>>(
        hdec_h, NH, hdWhh, NH, zbuf, nullptr, zbuf, NH4, NH);
    nmt_lstm_gate<<<(NB * NH) / 256, 256, 0, stream>>>(
        zbuf, c_dec, hF, hdec_h, (_Float16*)nullptr, 0);
    nmt_et<<<(NB * NS) / 128, 128, 0, stream>>>(e_proj, hF, e_buf);
    nmt_softmax<<<NB, 32, 0, stream>>>(e_buf, alpha);
    nmt_att<<<(NB * 3 * NH) / 256, 256, 0, stream>>>(alpha, enc_h, hdec_h, u_h);
    nmt_gemm<<<dim3(NH / 64, 1), 128, 0, stream>>>(
        u_h, 3 * NH, hWcomb, 3 * NH, nullptr, nullptr, Opre, NH, 3 * NH);
    nmt_tanh_h<<<(NB * NH) / 256, 256, 0, stream>>>(Opre, outs_h + (size_t)t * NB * NH, NB * NH);
  }

  // ---- 9. vocab projection with streaming logsumexp ----
  nmt_gemm_lse<<<dim3(NCH, MVOC / 64), 128, 0, stream>>>(outs_h, NH, hWvoc, NH, parts, NH);
  nmt_lse_reduce<<<MVOC, 32, 0, stream>>>(parts, NCH, lse);
  nmt_gold<<<MVOC, 32, 0, stream>>>(outs_h, hWvoc, tgt_pad, gold);
  nmt_final<<<1, NB, 0, stream>>>(gold, lse, tgt_pad, out);
}
